// ConLoss_798863917356
// MI455X (gfx1250) — compile-verified, matrix operand order checked by reading, not run on hardware
//
#include <hip/hip_runtime.h>
#include <hip/hip_bf16.h>
#include <math.h>

#define BDIM 2048
#define VV 2
#define DD 256
#define CC 1000
#define NN (VV*BDIM)                 // 4096
#define INV_T 14.285714285714286f    // 1/0.07 ; also the logsumexp shift (diag value)
#define CSPLIT 8
#define COLS_PER_CHUNK (NN / CSPLIT) // 512
#define CTILES (COLS_PER_CHUNK/16)   // 32

typedef __attribute__((ext_vector_type(16))) __bf16 v16bf;
typedef __attribute__((ext_vector_type(8)))  float  v8f;

// ---------------- kernel 0: zero the bincount array ----------------
__global__ void k_init(int* __restrict__ counts) {
    int i = blockIdx.x * blockDim.x + threadIdx.x;
    if (i < CC) counts[i] = 0;
}

// ---------------- kernel 1: per-sample softmax path ----------------
// rowmax -> e=exp(x-max); new_target = Y*e / sum(Y*e); argmax(Y*e) (tie -> lowest idx);
// P = Y*predicted_score; atomic bincount of argmax.
__global__ void k_prep(const float* __restrict__ outputs, const float* __restrict__ Y,
                       const float* __restrict__ ps, float* __restrict__ newt,
                       float* __restrict__ P, int* __restrict__ t, int* __restrict__ counts) {
    __shared__ float ru[CC];
    __shared__ float red[256];
    __shared__ float sval[256];
    __shared__ int   sidx[256];
    const int b = blockIdx.x, tid = threadIdx.x;
    const float* orow = outputs + (size_t)b * CC;
    const float* yrow = Y       + (size_t)b * CC;
    const float* prow = ps      + (size_t)b * CC;

    float lm = -3.4e38f;
    for (int c = tid; c < CC; c += 256) lm = fmaxf(lm, orow[c]);
    red[tid] = lm; __syncthreads();
    for (int s = 128; s > 0; s >>= 1) { if (tid < s) red[tid] = fmaxf(red[tid], red[tid + s]); __syncthreads(); }
    const float mx = red[0];
    __syncthreads();

    float ls = 0.f, bv = -1.f; int bi = 0;
    for (int c = tid; c < CC; c += 256) {
        float y = yrow[c];
        float e = y * __expf(orow[c] - mx);   // softmax denom cancels everywhere
        ru[c] = e; ls += e;
        if (e > bv) { bv = e; bi = c; }       // strided ascending -> first-max per thread
        P[(size_t)b * CC + c] = y * prow[c];
    }
    red[tid] = ls; sval[tid] = bv; sidx[tid] = bi; __syncthreads();
    for (int s = 128; s > 0; s >>= 1) {
        if (tid < s) {
            red[tid] += red[tid + s];
            float v2 = sval[tid + s]; int i2 = sidx[tid + s];
            if (v2 > sval[tid] || (v2 == sval[tid] && i2 < sidx[tid])) { sval[tid] = v2; sidx[tid] = i2; }
        }
        __syncthreads();
    }
    const float rsum = red[0];
    const int   am   = sidx[0];
    for (int c = tid; c < CC; c += 256) newt[(size_t)b * CC + c] = ru[c] / rsum;
    if (tid == 0) { t[b] = am; atomicAdd(&counts[am], 1); }
}

// ---------------- kernel 2: per-sample 1/count[t[b]] ----------------
__global__ void k_invc(const int* __restrict__ t, const int* __restrict__ counts,
                       float* __restrict__ invc) {
    int b = blockIdx.x * blockDim.x + threadIdx.x;
    if (b < BDIM) invc[b] = 1.0f / (float)counts[t[b]];
}

// ---------------- kernel 3: features [B,V,D] -> contrast bf16 [N,D] (view-major) ----
__global__ void k_conv(const float* __restrict__ feat, __bf16* __restrict__ cb) {
    const int n = blockIdx.x, tid = threadIdx.x; // 128 threads
    const int v = n / BDIM, bs = n % BDIM;
    const float* src = feat + ((size_t)bs * VV + v) * DD;
    __bf16* dst = cb + (size_t)n * DD;
    for (int d = tid; d < DD; d += 128) dst[d] = (__bf16)src[d];
}

// ---------------- kernel 4: WMMA GEMM + fused fixed-shift softmax epilogue --------
// One wave per 16-row tile; grid.y partitions columns into CSPLIT chunks.
// C-layout: element (M = v + 8*lanehi, N = lane&15) in acc[v].
// Fixed logsumexp shift M0 = 1/T (the diagonal value, provably >= all off-diag dots
// of unit-norm rows up to bf16 epsilon) -> branchless, one exp per element, no
// online-max bookkeeping. Mathematically identical: log_prob = adc - logsumexp(row).
__global__ void __launch_bounds__(128) k_gemm(const __bf16* __restrict__ cb,
                                              const float* __restrict__ P,
                                              const int*   __restrict__ t,
                                              const float* __restrict__ invc,
                                              float* __restrict__ partial) {
    const int lane  = threadIdx.x & 31;
    const int wave  = threadIdx.x >> 5;
    const int rt    = blockIdx.x * 4 + wave;
    const int chunk = blockIdx.y;
    const int ln    = lane & 15;
    const int hi    = lane >> 4;

    // A fragments for this wave's 16 rows (K=256 -> 8 frags, held in registers)
    const int rowA = rt * 16 + ln;
    v16bf a[8];
#pragma unroll
    for (int kk = 0; kk < 8; kk++)
        a[kk] = *(const v16bf*)(cb + (size_t)rowA * DD + kk * 32 + hi * 16);

    float ssum[8], smm[8], smadc[8];
#pragma unroll
    for (int v = 0; v < 8; v++) { ssum[v] = 0.f; smm[v] = 0.f; smadc[v] = 0.f; }

    const int colBase0 = chunk * COLS_PER_CHUNK;
    for (int ct = 0; ct < CTILES; ct++) {
        const int colAbs = colBase0 + ct * 16 + ln;   // this lane's column
        v8f acc;
#pragma unroll
        for (int v = 0; v < 8; v++) acc[v] = 0.f;
#pragma unroll
        for (int kk = 0; kk < 8; kk++) {
            v16bf bfrag = *(const v16bf*)(cb + (size_t)colAbs * DD + kk * 32 + hi * 16);
            acc = __builtin_amdgcn_wmma_f32_16x16x32_bf16(
                false, a[kk], false, bfrag, (short)0, acc, false, false);
        }
        const int   jb = colAbs & (BDIM - 1);
        const int   tc = t[jb];
        const float ic = invc[jb];
#pragma unroll
        for (int v = 0; v < 8; v++) {
            const int   i   = rt * 16 + v + hi * 8;
            const float val = acc[v] * INV_T;
            const float nd  = (i == colAbs) ? 0.f : 1.f;   // exclude self-contrast
            ssum[v] += nd * __expf(val - INV_T);           // fixed-shift exp-sum
            // partial-label mask: m[i,j] = P[i mod B, t[j mod B]] / count
            const int   ib = i & (BDIM - 1);
            const float m  = nd * P[(size_t)ib * CC + tc] * ic;
            smm[v]   += m;
            smadc[v] += m * val;
        }
    }

    // reduce across the 16 lanes of each half (rows v/v+8 live in separate halves)
#pragma unroll
    for (int v = 0; v < 8; v++) {
        float lss = ssum[v], lm = smm[v], lma = smadc[v];
        for (int off = 1; off < 16; off <<= 1) {
            lss += __shfl_xor(lss, off, 32);
            lm  += __shfl_xor(lm , off, 32);
            lma += __shfl_xor(lma, off, 32);
        }
        if (ln == 0) {
            const int r = v + hi * 8;
            float* pp = partial + (((size_t)rt * CSPLIT + chunk) * 16 + r) * 4;
            pp[0] = lss; pp[1] = lm; pp[2] = lma;
        }
    }
}

// ---------------- kernel 5: merge chunks per row, produce scalar loss ----------------
__global__ void k_final(const float* __restrict__ partial, float* __restrict__ out) {
    __shared__ float red[256];
    const int tid = threadIdx.x;
    float accl = 0.f;
    for (int i = tid; i < NN; i += 256) {
        const int rt = i >> 4, r = i & 15;
        float ss = 0.f, sm = 0.f, sma = 0.f;
        for (int c = 0; c < CSPLIT; c++) {
            const float* pp = partial + (((size_t)rt * CSPLIT + c) * 16 + r) * 4;
            ss += pp[0]; sm += pp[1]; sma += pp[2];
        }
        // mean_log_prob_pos = sum(m*adc) - logsumexp * sum(m),
        // logsumexp = M0 + log(sum exp(adc - M0)) with M0 = 1/T
        const float mean = sma - (INV_T + __logf(ss)) * sm;
        accl -= mean;   // loss contribution (T/baseT == 1)
    }
    red[tid] = accl; __syncthreads();
    for (int s = 128; s > 0; s >>= 1) { if (tid < s) red[tid] += red[tid + s]; __syncthreads(); }
    if (tid == 0) out[0] = red[0] / (float)NN;
}

extern "C" void kernel_launch(void* const* d_in, const int* in_sizes, int n_in,
                              void* d_out, int out_size, void* d_ws, size_t ws_size,
                              hipStream_t stream) {
    const float* outputs  = (const float*)d_in[0];   // [B, C]
    const float* features = (const float*)d_in[1];   // [B, V, D]
    const float* Y        = (const float*)d_in[2];   // [B, C]
    const float* ps       = (const float*)d_in[3];   // [B, C]
    float* out = (float*)d_out;                      // [1 + B*C] : loss, new_target

    char* W = (char*)d_ws;
    __bf16* cb      = (__bf16*)(W);                  //  2,097,152 B : bf16 contrast [N, D]
    float*  P       = (float*) (W +  2097152);       //  8,192,000 B : Y * predicted_score
    int*    t       = (int*)   (W + 10289152);       //      8,192 B : target_predict
    float*  invc    = (float*) (W + 10297344);       //      8,192 B : 1/count[t[b]]
    int*    counts  = (int*)   (W + 10305536);       //      4,096 B : bincount
    float*  partial = (float*) (W + 10309632);       //    524,288 B : per (rt,chunk,row) stats

    k_init<<<(CC + 255) / 256, 256, 0, stream>>>(counts);
    k_prep<<<BDIM, 256, 0, stream>>>(outputs, Y, ps, out + 1, P, t, counts);
    k_invc<<<BDIM / 256, 256, 0, stream>>>(t, counts, invc);
    k_conv<<<NN, 128, 0, stream>>>(features, cb);
    dim3 g(NN / 16 / 4, CSPLIT);
    k_gemm<<<g, 128, 0, stream>>>(cb, P, t, invc, partial);
    k_final<<<1, 256, 0, stream>>>(partial, out);
}
